// BWGNN_Hetero_14078902796340
// MI455X (gfx1250) — matrix operand order repaired and checked by reading, not compiled, not run on hardware
//
#include <hip/hip_runtime.h>
#include <hip/hip_bf16.h>

// ---------------------------------------------------------------------------
// BWGNN-Hetero on gfx1250 (MI455X, wave32, WMMA).
// Restructuring: all 3 beta-wavelet polynomials share the Krylov sequence
// f_k = L^k h, and the concat-GEMM factorizes as h_new = sum_k f_k @ Wtil_k^T + b3
// with Wtil_k = sum_j THETA[j][k] * W3[:, j*64:(j+1)*64].
// -> 2 edge-scatter passes per relation (6 total); GEMMs use bf16 WMMA
//    (v_wmma_f32_16x16x32_bf16, fp32 accum) fed from LDS-staged weights.
// Data movement: W staged global->LDS once per block via
// GLOBAL_LOAD_ASYNC_TO_LDS_B128 (ASYNCcnt / s_wait_asynccnt), then swizzled
// to bf16 wave32 B-fragment layout and read back with ds loads.
// ---------------------------------------------------------------------------

typedef __attribute__((ext_vector_type(16))) __bf16 v16bf;
typedef __attribute__((ext_vector_type(8)))  float  v8f;

#define HF 64           // hidden feature width
#define FLAG_LRELU 1
#define FLAG_ACCUM 2
#define FLAG_BIAS  4

// ---------------------------------------------------------------------------
// C[M,64] (op)= act( A[M,K] @ W[64,K]^T + bias )
// One wave computes a 16-row x 64-col output strip (4 WMMA N-tiles).
// Fragment layouts used here (wave32; same contiguous K order for A and B, not the ISA table):
//   A 16x32 bf16 : lane<16 -> row=lane,   K = k0 .. k0+15
//                  lane>=16 -> row=lane-16,K = k0+16 .. k0+31
//   B 32x16 bf16 : N = lane%16, K = k0 + (lane<16 ? 0 : 16) + i   (i=0..15)
//                  -> 16 *contiguous* K values of one W row (W is [Nout,K]).
//   C/D 16x16 f32: N = lane%16, M = vgpr + (lane<16 ? 0 : 8)
// Shared memory: [ K*64 fp32 raw W | (K/32)*4*32 fragments * 16 bf16 ]
// ---------------------------------------------------------------------------
__global__ void __launch_bounds__(256) gemm_bf16_wmma(
    const float* __restrict__ A, const float* __restrict__ W,
    const float* __restrict__ bias, float* __restrict__ C,
    int M, int K, int flags)
{
  extern __shared__ char smem[];
  float*  wF32  = (float*)smem;                          // [64][K] raw fp32 W
  __bf16* wFrag = (__bf16*)(smem + (size_t)HF * K * 4);  // pre-swizzled bf16 B frags

  // ---- stage W into LDS with async global->LDS copies (16B per lane) ----
  {
    unsigned lbase = (unsigned)(uintptr_t)wF32;  // low 32 bits = LDS offset
    int nchunk = K * 16;                         // (64*K*4B)/16B ; multiple of 256
    for (int c = threadIdx.x; c < nchunk; c += 256) {
      const float* g = W + (size_t)c * 4;
      unsigned l = lbase + (unsigned)c * 16u;
      asm volatile("global_load_async_to_lds_b128 %0, %1, off"
                   :: "v"(l), "v"(g) : "memory");
    }
    asm volatile("s_wait_asynccnt 0x0" ::: "memory");
    __syncthreads();

    // convert once to bf16 in exact per-lane B-fragment order:
    // frag id = ks*128 + t*32 + lane ; holds W[t*16 + lane%16][ks*32 + (lane>>4)*16 .. +15]
    int nfrag = (K >> 5) * 128;                  // multiple of 256 for K in {64,128}
    for (int fid = threadIdx.x; fid < nfrag; fid += 256) {
      int ks = fid >> 7, t = (fid >> 5) & 3, lane = fid & 31;
      int lrow = lane & 15, half = lane >> 4;
      const float* sp = wF32 + (size_t)(t * 16 + lrow) * K + (ks << 5) + half * 16;
      __bf16* dp = wFrag + (size_t)fid * 16;
#pragma unroll
      for (int i = 0; i < 16; ++i) dp[i] = (__bf16)sp[i];
    }
    __syncthreads();
  }

  int gid  = blockIdx.x * blockDim.x + threadIdx.x;
  int wave = gid >> 5;
  int lane = threadIdx.x & 31;
  int m0   = wave * 16;
  if (m0 >= M) return;                 // whole-wave uniform exit, after barriers
  int half = lane >> 4;
  int lrow = lane & 15;

  v8f zero = {};
  v8f acc[4];
#pragma unroll
  for (int t = 0; t < 4; ++t) acc[t] = zero;

  for (int k0 = 0; k0 < K; k0 += 32) {
    // ---- A fragment (convert fp32 -> bf16 in registers) ----
    const float* arow = A + (size_t)(m0 + lrow) * K + (k0 + half * 16);
    v16bf a;
#pragma unroll
    for (int i = 0; i < 8; ++i) {
      a[i]     = (__bf16)arow[i];       // K = k0 + half*16 + i
      a[8 + i] = (__bf16)arow[8 + i];   // K = k0 + half*16 + 8 + i
    }
    // ---- 4 N-tiles: B fragments straight from LDS ----
    int ks = k0 >> 5;
#pragma unroll
    for (int t = 0; t < 4; ++t) {
      v16bf b = *(const v16bf*)(wFrag + (size_t)(ks * 128 + t * 32 + lane) * 16);
      acc[t] = __builtin_amdgcn_wmma_f32_16x16x32_bf16(
          /*neg_a=*/false, a, /*neg_b=*/false, b,
          /*c_mod=*/(short)0, acc[t], /*reuse_a=*/false, /*reuse_b=*/false);
    }
  }

  // ---- epilogue: bias / leaky-relu / (accumulate|store) ----
#pragma unroll
  for (int t = 0; t < 4; ++t) {
    int col = t * 16 + lrow;
    float bv = (flags & FLAG_BIAS) ? bias[col] : 0.0f;
#pragma unroll
    for (int d = 0; d < 8; ++d) {
      int row = m0 + half * 8 + d;
      float v = acc[t][d] + bv;
      if (flags & FLAG_LRELU) v = (v > 0.0f) ? v : 0.01f * v;
      float* cp = C + (size_t)row * HF + col;
      if (flags & FLAG_ACCUM) *cp += v; else *cp = v;
    }
  }
}

// Wtil[k][n][c] = sum_j THETA[j][k] * W3[n][j*64 + c]   (THETA for d=2 wavelets)
__global__ void wtil_kernel(const float* __restrict__ W3, float* __restrict__ Wtil)
{
  int i = blockIdx.x * blockDim.x + threadIdx.x;
  if (i >= 3 * HF * HF) return;
  int k = i >> 12, n = (i >> 6) & 63, c = i & 63;
  float t0, t1, t2;                      // THETA[j][k], j = 0..2
  if (k == 0)      { t0 = 3.0f;  t1 = 0.0f;  t2 = 0.0f;  }
  else if (k == 1) { t0 = -3.0f; t1 = 3.0f;  t2 = 0.0f;  }
  else             { t0 = 0.75f; t1 = -1.5f; t2 = 0.75f; }
  const float* r = W3 + (size_t)n * (3 * HF) + c;
  Wtil[i] = t0 * r[0] + t1 * r[HF] + t2 * r[2 * HF];
}

__global__ void degree_kernel(const int* __restrict__ dst, float* __restrict__ deg, int E)
{
  int e = blockIdx.x * blockDim.x + threadIdx.x;
  if (e < E) atomicAdd(&deg[dst[e]], 1.0f);
}

__global__ void dinv_kernel(const float* __restrict__ deg, float* __restrict__ dinv, int n)
{
  int i = blockIdx.x * blockDim.x + threadIdx.x;
  if (i < n) dinv[i] = rsqrtf(fmaxf(deg[i], 1.0f));
}

// agg[dst[e]][:] += f[src[e]][:] * dinv[src[e]]   — 1 wave per edge, float2/lane
__global__ void __launch_bounds__(256) scatter_kernel(
    const int* __restrict__ src, const int* __restrict__ dst,
    const float* __restrict__ dinv, const float* __restrict__ fin,
    float* __restrict__ agg, int E)
{
  int gid  = blockIdx.x * blockDim.x + threadIdx.x;
  int e    = gid >> 5;
  int lane = gid & 31;
  if (e >= E) return;
  int s = src[e], d = dst[e];
  float ds = dinv[s];
  float2 v = ((const float2*)(fin + (size_t)s * HF))[lane];
  float* ap = agg + (size_t)d * HF + 2 * lane;
  atomicAdd(ap,     v.x * ds);
  atomicAdd(ap + 1, v.y * ds);
}

// fout = fin - agg * dinv[row]
__global__ void update_kernel(const float* __restrict__ fin, const float* __restrict__ agg,
                              const float* __restrict__ dinv, float* __restrict__ fout, int total)
{
  int i = blockIdx.x * blockDim.x + threadIdx.x;
  if (i < total) fout[i] = fin[i] - agg[i] * dinv[i >> 6];
}

__global__ void accum_kernel(float* __restrict__ acc, const float* __restrict__ x, int total)
{
  int i = blockIdx.x * blockDim.x + threadIdx.x;
  if (i < total) acc[i] += x[i];
}

// out[n][co] = sum_c lrelu(h_all[n][c]) * W4[co][c] + b4[co]  (C=2)
__global__ void final_kernel(const float* __restrict__ hall, const float* __restrict__ W4,
                             const float* __restrict__ b4, float* __restrict__ out, int n)
{
  int i = blockIdx.x * blockDim.x + threadIdx.x;
  if (i >= n) return;
  const float* hr = hall + (size_t)i * HF;
  float a0 = b4[0], a1 = b4[1];
#pragma unroll 8
  for (int c = 0; c < HF; ++c) {
    float v = hr[c];
    v = (v > 0.0f) ? v : 0.01f * v;
    a0 += v * W4[c];
    a1 += v * W4[HF + c];
  }
  out[2 * i]     = a0;
  out[2 * i + 1] = a1;
}

// ---------------------------------------------------------------------------
extern "C" void kernel_launch(void* const* d_in, const int* in_sizes, int n_in,
                              void* d_out, int out_size, void* d_ws, size_t ws_size,
                              hipStream_t stream)
{
  const float* in_feat = (const float*)d_in[0];
  const float* W1 = (const float*)d_in[1];
  const float* b1 = (const float*)d_in[2];
  const float* W2 = (const float*)d_in[3];
  const float* b2 = (const float*)d_in[4];
  const float* W3 = (const float*)d_in[5];
  const float* b3 = (const float*)d_in[6];
  const float* W4 = (const float*)d_in[7];
  const float* b4 = (const float*)d_in[8];
  const int* srcs[3] = {(const int*)d_in[9],  (const int*)d_in[11], (const int*)d_in[13]};
  const int* dsts[3] = {(const int*)d_in[10], (const int*)d_in[12], (const int*)d_in[14]};

  const int IN = 128;
  const int N  = in_sizes[0] / IN;     // 100000
  const int E  = in_sizes[9];          // 1600000

  // workspace layout (floats)
  float* ws   = (float*)d_ws;
  size_t nh   = (size_t)N * HF;
  float* hA   = ws;                    // node features (ping)
  float* hB   = hA + nh;               // node features (pong)
  float* f    = hB + nh;               // Krylov vector f_k
  float* agg  = f + nh;                // scatter accumulator
  float* hall = agg + nh;              // h_all
  float* deg  = hall + nh;             // [N]
  float* dinv = deg + N;               // [N]
  float* Wtil = dinv + N;              // [3][64][64]

  const int TB = 256;
  dim3 blk(TB);
  int gemm_blocks = (int)(((size_t)(N / 16) * 32 + TB - 1) / TB);
  int ew_blocks   = (int)((nh + TB - 1) / TB);             // elementwise N*64
  int node_blocks = (N + TB - 1) / TB;
  int edge_blocks = (E + TB - 1) / TB;
  int scat_blocks = (int)(((size_t)E * 32 + TB - 1) / TB);
  size_t sh128 = (size_t)HF * IN * 6;  // raw fp32 + bf16 frags, K=128 -> 48KB
  size_t sh64  = (size_t)HF * HF * 6;  // K=64 -> 24KB

  // 0) fold THETAS into W3
  wtil_kernel<<<(3 * HF * HF + TB - 1) / TB, blk, 0, stream>>>(W3, Wtil);

  // 1) h1 = lrelu(in_feat @ W1^T + b1)   -> f (temp)
  gemm_bf16_wmma<<<gemm_blocks, blk, sh128, stream>>>(in_feat, W1, b1, f, N, IN,
                                                      FLAG_LRELU | FLAG_BIAS);
  // 2) h = lrelu(h1 @ W2^T + b2)         -> hA
  gemm_bf16_wmma<<<gemm_blocks, blk, sh64, stream>>>(f, W2, b2, hA, N, HF,
                                                     FLAG_LRELU | FLAG_BIAS);

  hipMemsetAsync(hall, 0, nh * sizeof(float), stream);

  float* h_cur = hA;
  float* h_nxt = hB;
  for (int r = 0; r < 3; ++r) {
    const int* src = srcs[r];
    const int* dst = dsts[r];

    // degree / d^{-1/2}
    hipMemsetAsync(deg, 0, (size_t)N * sizeof(float), stream);
    degree_kernel<<<edge_blocks, blk, 0, stream>>>(dst, deg, E);
    dinv_kernel<<<node_blocks, blk, 0, stream>>>(deg, dinv, N);

    // k = 0 : h_nxt = h_cur @ Wtil0^T + b3
    gemm_bf16_wmma<<<gemm_blocks, blk, sh64, stream>>>(h_cur, Wtil, b3, h_nxt,
                                                       N, HF, FLAG_BIAS);

    // k = 1 : f = L h_cur ;  h_nxt += f @ Wtil1^T
    hipMemsetAsync(agg, 0, nh * sizeof(float), stream);
    scatter_kernel<<<scat_blocks, blk, 0, stream>>>(src, dst, dinv, h_cur, agg, E);
    update_kernel<<<ew_blocks, blk, 0, stream>>>(h_cur, agg, dinv, f, (int)nh);
    gemm_bf16_wmma<<<gemm_blocks, blk, sh64, stream>>>(f, Wtil + 1 * HF * HF, b3, h_nxt,
                                                       N, HF, FLAG_ACCUM);

    // k = 2 : f = L f (in place) ;  h_nxt += f @ Wtil2^T
    hipMemsetAsync(agg, 0, nh * sizeof(float), stream);
    scatter_kernel<<<scat_blocks, blk, 0, stream>>>(src, dst, dinv, f, agg, E);
    update_kernel<<<ew_blocks, blk, 0, stream>>>(f, agg, dinv, f, (int)nh);
    gemm_bf16_wmma<<<gemm_blocks, blk, sh64, stream>>>(f, Wtil + 2 * HF * HF, b3, h_nxt,
                                                       N, HF, FLAG_ACCUM);

    // h_all += h ; swap ping-pong
    accum_kernel<<<ew_blocks, blk, 0, stream>>>(hall, h_nxt, (int)nh);
    float* t = h_cur; h_cur = h_nxt; h_nxt = t;
  }

  // out = lrelu(h_all) @ W4^T + b4
  final_kernel<<<node_blocks, blk, 0, stream>>>(hall, W4, b4, (float*)d_out, N);
}
